// VisualEncoder_6296422056355
// MI455X (gfx1250) — matrix operand order repaired and checked by reading, not compiled, not run on hardware
//
#include <hip/hip_runtime.h>
#include <hip/hip_bf16.h>

typedef __bf16 bf16;
typedef __attribute__((ext_vector_type(16))) bf16  v16bf;
typedef __attribute__((ext_vector_type(8)))  bf16  v8bf;
typedef __attribute__((ext_vector_type(8)))  float v8f;
typedef int v4i __attribute__((vector_size(16)));   // matches builtin param type

// ---------------- problem constants ----------------
constexpr int kD[4] = {8, 4, 2, 1};
constexpr int kH[4] = {24, 12, 6, 3};
constexpr int kW[4] = {24, 12, 6, 3};
constexpr int kStart[4] = {0, 4608, 5184, 5256};
constexpr int LEN = 5265;            // 4608+576+72+9
constexpr int BATCH = 8;             // bs*nf = 2*4
constexpr int MROWS = BATCH * LEN;   // 42120
constexpr int DM = 256;
constexpr int NH = 8;
constexpr int HD = 32;
constexpr int DFFN = 1024;
constexpr int NLAYERS = 6;

// ---------------- CDNA5 async global->LDS path (probe via __has_builtin) ----
#if defined(__gfx1250__) && __has_builtin(__builtin_amdgcn_global_load_async_to_lds_b128)
#define HAS_ASYNC_LDS 1
#else
#define HAS_ASYNC_LDS 0
#endif

#if HAS_ASYNC_LDS
#define ASYNC_B128(gp, lp)                                                \
  __builtin_amdgcn_global_load_async_to_lds_b128(                         \
      (__attribute__((address_space(1))) v4i*)(v4i*)(void*)(gp),          \
      (__attribute__((address_space(3))) v4i*)(v4i*)(void*)(lp), 0, 0)
#if __has_builtin(__builtin_amdgcn_s_wait_asynccnt)
#define ASYNC_WAIT(n) __builtin_amdgcn_s_wait_asynccnt(n)
#else
#define ASYNC_WAIT(n) asm volatile("s_wait_asynccnt " #n ::: "memory")
#endif
#else
#define ASYNC_WAIT(n) ((void)0)
#endif

// ================= pack inputs =================
__global__ __launch_bounds__(256) void pack_kernel(
    const float* __restrict__ s0, const float* __restrict__ s1,
    const float* __restrict__ s2, const float* __restrict__ s3,
    const float* __restrict__ p0, const float* __restrict__ p1,
    const float* __restrict__ p2, const float* __restrict__ p3,
    const float* __restrict__ le,
    float* __restrict__ x, float* __restrict__ pos) {
  size_t i = (size_t)blockIdx.x * 256 + threadIdx.x;
  size_t total = (size_t)MROWS * DM;
  if (i >= total) return;
  int c = (int)(i % DM);
  size_t t = i / DM;
  int n = (int)(t % LEN);
  int bf = (int)(t / LEN);
  int lv, s, nl;
  const float* sp;
  const float* pp;
  if (n < 4608)      { lv = 0; s = n;        nl = 4608; sp = s0; pp = p0; }
  else if (n < 5184) { lv = 1; s = n - 4608; nl = 576;  sp = s1; pp = p1; }
  else if (n < 5256) { lv = 2; s = n - 5184; nl = 72;   sp = s2; pp = p2; }
  else               { lv = 3; s = n - 5256; nl = 9;    sp = s3; pp = p3; }
  size_t si = ((size_t)bf * DM + c) * nl + s;
  x[i] = sp[si];
  pos[i] = pp[si] + le[lv * DM + c];
}

__global__ void make_ref(float* __restrict__ ref) {
  int q = blockIdx.x * blockDim.x + threadIdx.x;
  if (q >= LEN) return;
  int lv, s;
  if (q < 4608)      { lv = 0; s = q; }
  else if (q < 5184) { lv = 1; s = q - 4608; }
  else if (q < 5256) { lv = 2; s = q - 5184; }
  else               { lv = 3; s = q - 5256; }
  int D = kD[lv], H = kH[lv], W = kW[lv];
  int xi = s % W, yi = (s / W) % H, zi = s / (W * H);
  ref[q * 3 + 0] = (xi + 0.5f) / (float)W;
  ref[q * 3 + 1] = (yi + 0.5f) / (float)H;
  ref[q * 3 + 2] = (zi + 0.5f) / (float)D;
}

// ================= elementwise helpers =================
__global__ __launch_bounds__(256) void to_bf16_kernel(const float* __restrict__ a,
                                                      bf16* __restrict__ o, size_t n) {
  size_t i = (size_t)blockIdx.x * 256 + threadIdx.x;
  if (i < n) o[i] = (bf16)a[i];
}
__global__ __launch_bounds__(256) void add_to_bf16_kernel(const float* __restrict__ a,
                                                          const float* __restrict__ b,
                                                          bf16* __restrict__ o, size_t n) {
  size_t i = (size_t)blockIdx.x * 256 + threadIdx.x;
  if (i < n) o[i] = (bf16)(a[i] + b[i]);
}
__global__ __launch_bounds__(256) void copy_f32_kernel(const float* __restrict__ a,
                                                       float* __restrict__ o, size_t n) {
  size_t i = (size_t)blockIdx.x * 256 + threadIdx.x;
  if (i < n) o[i] = a[i];
}
// transpose + convert weight (K x N f32 row-major) -> Wt (N x K bf16 row-major)
__global__ __launch_bounds__(256) void wtrans_kernel(const float* __restrict__ Wkn,
                                                     bf16* __restrict__ Wt, int K, int N) {
  int id = blockIdx.x * 256 + threadIdx.x;
  if (id >= K * N) return;
  int n = id / K, k = id % K;
  Wt[id] = (bf16)Wkn[(size_t)k * N + n];
}

// ================= WMMA GEMM =================
// C[M,N] = A[M,K](bf16) * Bt[N,K]^T(bf16) + bias ; EPI: 0=bias, 1=bias+relu
// Block tile 64x128, 8 waves (2x4), each wave a 32x32 subtile (2x2 wmma tiles).
// Double-buffered LDS staging via async global->LDS when available.
// Out-of-range A rows are CLAMPED (not zeroed): they only affect C rows >= M,
// which are never stored, so clamping is correct and branch-free.
template <int EPI, bool OUT_BF16>
__global__ __launch_bounds__(256) void gemm_wmma(
    const bf16* __restrict__ A, const bf16* __restrict__ Bt,
    const float* __restrict__ bias, void* __restrict__ Out,
    int Mr, int N, int K) {
  constexpr int BM = 64, BN = 128, BK = 32, LDA = 40, LDB = 40;
  __shared__ bf16 lsA[2][BM * LDA];
  __shared__ bf16 lsB[2][BN * LDB];

  const int tid = threadIdx.x;
  const int lane = tid & 31;
  const int wave = tid >> 5;
  const int waveM = wave & 1;   // 0..1 -> 32-row slice
  const int waveN = wave >> 1;  // 0..3 -> 32-col slice
  const int blockN0 = blockIdx.x * BN;
  const int blockM0 = blockIdx.y * BM;

  const int m16 = lane & 15;
  const int half = lane >> 4;       // 0/1
  const int aklo = half * 8;        // A frag k-offset (ISA 16-bit A 16x32 layout)
  const int bkhalf = half * 8;      // B frag k-offset (same layout as A)

  const int lr = tid >> 2;          // 0..63 tile row
  const int lc = (tid & 3) * 8;     // 0,8,16,24 tile col (8 bf16 = 16B)

  int garow = blockM0 + lr;
  if (garow >= Mr) garow = Mr - 1;  // clamp (see note above)
  const bf16* gA  = A  + (size_t)garow * K + lc;
  const bf16* gB0 = Bt + (size_t)(blockN0 + lr) * K + lc;
  const bf16* gB1 = Bt + (size_t)(blockN0 + lr + 64) * K + lc;
  bf16* sA  = &lsA[0][0] + lr * LDA + lc;
  bf16* sB0 = &lsB[0][0] + lr * LDB + lc;
  bf16* sB1 = &lsB[0][0] + (lr + 64) * LDB + lc;

  auto stage = [&](int buf, int k0) {
#if HAS_ASYNC_LDS
    ASYNC_B128(gA + k0,  sA  + buf * (BM * LDA));
    ASYNC_B128(gB0 + k0, sB0 + buf * (BN * LDB));
    ASYNC_B128(gB1 + k0, sB1 + buf * (BN * LDB));
#else
    v8bf a  = *(const v8bf*)(gA + k0);
    v8bf b0 = *(const v8bf*)(gB0 + k0);
    v8bf b1 = *(const v8bf*)(gB1 + k0);
    *(v8bf*)(sA + buf * (BM * LDA)) = a;
    *(v8bf*)(sB0 + buf * (BN * LDB)) = b0;
    *(v8bf*)(sB1 + buf * (BN * LDB)) = b1;
    if (k0 + BK < K) {  // prefetch next tile -> global_prefetch_b8
      __builtin_prefetch(gA + k0 + BK, 0, 1);
      __builtin_prefetch(gB0 + k0 + BK, 0, 1);
    }
#endif
  };

  v8f acc[2][2] = {};
  const int nk = K / BK;
  stage(0, 0);
  for (int ks = 0; ks < nk; ++ks) {
    const int buf = ks & 1;
    if (ks + 1 < nk) {
      stage(buf ^ 1, (ks + 1) * BK);  // issue next stage (other buffer)
      ASYNC_WAIT(3);                  // current stage done; next (3 ops) in flight
    } else {
      ASYNC_WAIT(0);
    }
    __syncthreads();

    v16bf afr[2], bfr[2];
#pragma unroll
    for (int mt = 0; mt < 2; ++mt) {
      const bf16* p = &lsA[buf][(waveM * 32 + mt * 16 + m16) * LDA + aklo];
      v8bf lo = *(const v8bf*)p;
      v8bf hi = *(const v8bf*)(p + 16);
#pragma unroll
      for (int i = 0; i < 8; ++i) { afr[mt][i] = lo[i]; afr[mt][8 + i] = hi[i]; }
    }
#pragma unroll
    for (int nt = 0; nt < 2; ++nt) {
      const bf16* p = &lsB[buf][(waveN * 32 + nt * 16 + m16) * LDB + bkhalf];
      v8bf lo = *(const v8bf*)p;
      v8bf hi = *(const v8bf*)(p + 16);
#pragma unroll
      for (int i = 0; i < 8; ++i) { bfr[nt][i] = lo[i]; bfr[nt][8 + i] = hi[i]; }
    }
#pragma unroll
    for (int mt = 0; mt < 2; ++mt)
#pragma unroll
      for (int nt = 0; nt < 2; ++nt)
        acc[mt][nt] = __builtin_amdgcn_wmma_f32_16x16x32_bf16(
            false, afr[mt], false, bfr[nt], (short)0, acc[mt][nt], false, false);
    __syncthreads();  // buffer `buf` is rewritten by stage ks+2 (issued next iter)
  }

  // epilogue: VGPR r, lane l -> m = r + (l>>4)*8, n = l%16
#pragma unroll
  for (int mt = 0; mt < 2; ++mt) {
#pragma unroll
    for (int nt = 0; nt < 2; ++nt) {
      int gn = blockN0 + waveN * 32 + nt * 16 + m16;
      float bv = bias[gn];
#pragma unroll
      for (int r = 0; r < 8; ++r) {
        int gm = blockM0 + waveM * 32 + mt * 16 + r + half * 8;
        if (gm < Mr) {
          float v = acc[mt][nt][r] + bv;
          if (EPI == 1) v = v > 0.f ? v : 0.f;
          if (OUT_BF16) ((bf16*)Out)[(size_t)gm * N + gn] = (bf16)v;
          else          ((float*)Out)[(size_t)gm * N + gn] = v;
        }
      }
    }
  }
}

// ================= softmax over 16 points/head =================
__global__ __launch_bounds__(256) void softmax16_kernel(float* __restrict__ attn, int total) {
  int t = blockIdx.x * 256 + threadIdx.x;
  if (t >= total) return;
  float* p = attn + (size_t)t * 16;
  float mx = p[0];
#pragma unroll
  for (int i = 1; i < 16; ++i) mx = fmaxf(mx, p[i]);
  float s = 0.f;
#pragma unroll
  for (int i = 0; i < 16; ++i) { float e = __expf(p[i] - mx); p[i] = e; s += e; }
  float inv = 1.f / s;
#pragma unroll
  for (int i = 0; i < 16; ++i) p[i] *= inv;
}

// ================= deformable sampling =================
// block = one (b,q); wave w = head w; lane = channel. value: (B,Len,8,32) f32.
__global__ __launch_bounds__(256) void msda_sample_kernel(
    const float* __restrict__ value, const float* __restrict__ off,
    const float* __restrict__ attn, const float* __restrict__ ref,
    bf16* __restrict__ outb) {
  int bq = blockIdx.x;                 // b*LEN + q
  int h = threadIdx.x >> 5;
  int c = threadIdx.x & 31;
  int b = bq / LEN;
  int q = bq - b * LEN;
  const float rx = ref[q * 3 + 0];
  const float ry = ref[q * 3 + 1];
  const float rz = ref[q * 3 + 2];
  const float* offp = off + (size_t)bq * 384 + h * 48;   // (lv,p,3)
  const float* atp  = attn + (size_t)bq * 128 + h * 16;  // (lv,p)
  const float* vbase = value + (size_t)b * LEN * DM + h * HD + c;
  float acc = 0.f;
#pragma unroll
  for (int lv = 0; lv < 4; ++lv) {
    const int D = kD[lv], H = kH[lv], W = kW[lv], start = kStart[lv];
    const float* vlev = vbase + (size_t)start * DM;
#pragma unroll
    for (int p = 0; p < 4; ++p) {
      float a  = atp[lv * 4 + p];
      float ox = offp[(lv * 4 + p) * 3 + 0];
      float oy = offp[(lv * 4 + p) * 3 + 1];
      float oz = offp[(lv * 4 + p) * 3 + 2];
      float fx = (rx + ox / (float)W) * W - 0.5f;
      float fy = (ry + oy / (float)H) * H - 0.5f;
      float fz = (rz + oz / (float)D) * D - 0.5f;
      float x0f = floorf(fx), y0f = floorf(fy), z0f = floorf(fz);
      int ix0 = (int)x0f, iy0 = (int)y0f, iz0 = (int)z0f;
      float wx1 = fx - x0f, wy1 = fy - y0f, wz1 = fz - z0f;
#pragma unroll
      for (int dz = 0; dz < 2; ++dz)
#pragma unroll
        for (int dy = 0; dy < 2; ++dy)
#pragma unroll
          for (int dx = 0; dx < 2; ++dx) {
            int xi = ix0 + dx, yi = iy0 + dy, zi = iz0 + dz;
            bool valid = (xi >= 0) & (xi < W) & (yi >= 0) & (yi < H) &
                         (zi >= 0) & (zi < D);
            if (valid) {
              float w = (dx ? wx1 : 1.f - wx1) * (dy ? wy1 : 1.f - wy1) *
                        (dz ? wz1 : 1.f - wz1);
              int idx = (zi * H + yi) * W + xi;
              acc += a * w * vlev[(size_t)idx * DM];
            }
          }
    }
  }
  outb[(size_t)bq * DM + h * HD + c] = (bf16)acc;
}

// ================= residual + LayerNorm (writes f32 x and bf16 copy) =============
__global__ __launch_bounds__(256) void add_ln_kernel(
    float* __restrict__ x, const float* __restrict__ a,
    const float* __restrict__ g, const float* __restrict__ be,
    bf16* __restrict__ xb) {
  __shared__ float s1[256];
  __shared__ float s2[256];
  int row = blockIdx.x, c = threadIdx.x;
  size_t i = (size_t)row * DM + c;
  float v = x[i] + a[i];
  s1[c] = v; s2[c] = v * v;
  __syncthreads();
  for (int st = 128; st > 0; st >>= 1) {
    if (c < st) { s1[c] += s1[c + st]; s2[c] += s2[c + st]; }
    __syncthreads();
  }
  float mean = s1[0] * (1.f / 256.f);
  float var = s2[0] * (1.f / 256.f) - mean * mean;
  float r = rsqrtf(var + 1e-5f);
  float y = (v - mean) * r * g[c] + be[c];
  x[i] = y;
  xb[i] = (bf16)y;
}

// ================= launcher =================
extern "C" void kernel_launch(void* const* d_in, const int* in_sizes, int n_in,
                              void* d_out, int out_size, void* d_ws, size_t ws_size,
                              hipStream_t stream) {
  (void)in_sizes; (void)n_in; (void)out_size; (void)ws_size;
  const float* src[4] = {(const float*)d_in[0], (const float*)d_in[2],
                         (const float*)d_in[4], (const float*)d_in[6]};
  const float* pin[4] = {(const float*)d_in[1], (const float*)d_in[3],
                         (const float*)d_in[5], (const float*)d_in[7]};
  const float* le   = (const float*)d_in[8];
  const float* Woff = (const float*)d_in[9];
  const float* boff = (const float*)d_in[10];
  const float* Wat  = (const float*)d_in[11];
  const float* bat  = (const float*)d_in[12];
  const float* Wv   = (const float*)d_in[13];
  const float* bv   = (const float*)d_in[14];
  const float* Wo   = (const float*)d_in[15];
  const float* bo   = (const float*)d_in[16];
  const float* g1   = (const float*)d_in[17];
  const float* be1  = (const float*)d_in[18];
  const float* W1   = (const float*)d_in[19];
  const float* b1   = (const float*)d_in[20];
  const float* W2   = (const float*)d_in[21];
  const float* b2   = (const float*)d_in[22];
  const float* g2   = (const float*)d_in[23];
  const float* be2  = (const float*)d_in[24];

  // ---- workspace layout ----
  char* ws = (char*)d_ws;
  size_t off = 0;
  auto alloc = [&](size_t bytes) {
    size_t o = off; off += (bytes + 255) & ~(size_t)255; return (void*)(ws + o);
  };
  const size_t M = MROWS;
  float* x_f   = (float*)alloc(M * DM * 4);
  float* pos_f = (float*)alloc(M * DM * 4);
  float* ref_f = (float*)alloc((size_t)LEN * 3 * 4);
  bf16*  xb    = (bf16*)alloc(M * DM * 2);
  bf16*  qb    = (bf16*)alloc(M * DM * 2);
  float* val_f = (float*)alloc(M * DM * 4);
  float* off_f = (float*)alloc(M * 384 * 4);
  float* att_f = (float*)alloc(M * 128 * 4);
  bf16*  msdab = (bf16*)alloc(M * DM * 2);
  float* aout  = (float*)alloc(M * DM * 4);
  bf16*  hidb  = (bf16*)alloc(M * DFFN * 2);
  bf16*  wv_t  = (bf16*)alloc(256 * 256 * 2);
  bf16*  woff_t= (bf16*)alloc(256 * 384 * 2);
  bf16*  wat_t = (bf16*)alloc(256 * 128 * 2);
  bf16*  wo_t  = (bf16*)alloc(256 * 256 * 2);
  bf16*  w1_t  = (bf16*)alloc(256 * 1024 * 2);
  bf16*  w2_t  = (bf16*)alloc(1024 * 256 * 2);

  const size_t nelem = M * DM;
  const int gElem = (int)((nelem + 255) / 256);
  const int gM64 = (int)((M + 63) / 64);

  pack_kernel<<<gElem, 256, 0, stream>>>(src[0], src[1], src[2], src[3],
                                         pin[0], pin[1], pin[2], pin[3],
                                         le, x_f, pos_f);
  make_ref<<<(LEN + 255) / 256, 256, 0, stream>>>(ref_f);
  to_bf16_kernel<<<gElem, 256, 0, stream>>>(x_f, xb, nelem);

  for (int l = 0; l < NLAYERS; ++l) {
    const float* Wv_l = Wv + (size_t)l * 256 * 256;   const float* bv_l = bv + l * 256;
    const float* Woff_l = Woff + (size_t)l * 256 * 384; const float* boff_l = boff + l * 384;
    const float* Wat_l = Wat + (size_t)l * 256 * 128; const float* bat_l = bat + l * 128;
    const float* Wo_l = Wo + (size_t)l * 256 * 256;   const float* bo_l = bo + l * 256;
    const float* W1_l = W1 + (size_t)l * 256 * 1024;  const float* b1_l = b1 + l * 1024;
    const float* W2_l = W2 + (size_t)l * 1024 * 256;  const float* b2_l = b2 + l * 256;

    wtrans_kernel<<<(256 * 256 + 255) / 256, 256, 0, stream>>>(Wv_l, wv_t, 256, 256);
    wtrans_kernel<<<(256 * 384 + 255) / 256, 256, 0, stream>>>(Woff_l, woff_t, 256, 384);
    wtrans_kernel<<<(256 * 128 + 255) / 256, 256, 0, stream>>>(Wat_l, wat_t, 256, 128);
    wtrans_kernel<<<(256 * 256 + 255) / 256, 256, 0, stream>>>(Wo_l, wo_t, 256, 256);
    wtrans_kernel<<<(256 * 1024 + 255) / 256, 256, 0, stream>>>(W1_l, w1_t, 256, 1024);
    wtrans_kernel<<<(1024 * 256 + 255) / 256, 256, 0, stream>>>(W2_l, w2_t, 1024, 256);

    // q = x + pos (bf16)
    add_to_bf16_kernel<<<gElem, 256, 0, stream>>>(x_f, pos_f, qb, nelem);

    gemm_wmma<0, false><<<dim3(2, gM64), 256, 0, stream>>>(xb, wv_t, bv_l, val_f, (int)M, 256, 256);
    gemm_wmma<0, false><<<dim3(3, gM64), 256, 0, stream>>>(qb, woff_t, boff_l, off_f, (int)M, 384, 256);
    gemm_wmma<0, false><<<dim3(1, gM64), 256, 0, stream>>>(qb, wat_t, bat_l, att_f, (int)M, 128, 256);
    softmax16_kernel<<<(int)((M * NH + 255) / 256), 256, 0, stream>>>(att_f, (int)(M * NH));
    msda_sample_kernel<<<(int)M, 256, 0, stream>>>(val_f, off_f, att_f, ref_f, msdab);
    gemm_wmma<0, false><<<dim3(2, gM64), 256, 0, stream>>>(msdab, wo_t, bo_l, aout, (int)M, 256, 256);
    add_ln_kernel<<<(int)M, 256, 0, stream>>>(x_f, aout, g1 + l * 256, be1 + l * 256, xb);

    gemm_wmma<1, true><<<dim3(8, gM64), 256, 0, stream>>>(xb, w1_t, b1_l, hidb, (int)M, 1024, 256);
    gemm_wmma<0, false><<<dim3(2, gM64), 256, 0, stream>>>(hidb, w2_t, b2_l, aout, (int)M, 256, 1024);
    add_ln_kernel<<<(int)M, 256, 0, stream>>>(x_f, aout, g2 + l * 256, be2 + l * 256, xb);
  }

  copy_f32_kernel<<<gElem, 256, 0, stream>>>(x_f, (float*)d_out, nelem);
}
